// SelfReferentialField_51891794870689
// MI455X (gfx1250) — matrix operand order repaired and checked by reading, not compiled, not run on hardware
//
#include <hip/hip_runtime.h>

// ---------------------------------------------------------------------------
// Types
// ---------------------------------------------------------------------------
typedef _Float16 h8  __attribute__((ext_vector_type(8)));
typedef _Float16 h16 __attribute__((ext_vector_type(16)));
typedef float    f8  __attribute__((ext_vector_type(8)));

#define BATCH   2048
#define IN_DIM  1024
#define HID_DIM 2048
#define OUT_DIM 1024
#define N_EXP_A 12
#define TOPK_A  4
#define N_EXP_G 6
#define H_TARGET 1.0114042647073518f

__device__ __forceinline__ h16 cat16(h8 a, h8 b) {
    return __builtin_shufflevector(a, b, 0,1,2,3,4,5,6,7,8,9,10,11,12,13,14,15);
}

__device__ __forceinline__ f8 wmma16(h16 a, h16 b, f8 c) {
    // D = A(16x32 f16) * B(32x16 f16) + C(16x16 f32)
    return __builtin_amdgcn_wmma_f32_16x16x32_f16(false, a, false, b, (short)0, c, false, false);
}

// ---------------------------------------------------------------------------
// Generic WMMA GEMM:  C = A(f16,[M,K],lda) @ B(f32,[K,N],ldb)  + epilogue
// Block tile 128(M) x 128(N), K-step 32, 256 threads = 8 waves (4x2).
// Each wave owns 32(M) x 64(N) = 2x4 v_wmma_f32_16x16x32_f16 tiles.
// A (f16 in global) is staged to LDS with global_load_async_to_lds_b128
// (ASYNCcnt); B (f32) is converted to f16 in VGPRs and stored transposed.
// ---------------------------------------------------------------------------
struct GemmP {
    const _Float16* A;  const float* B;  const float* bias;
    int M, N, K, lda, ldb;
    _Float16* outH;                 // EPI 0 (relu) / 2 (tanh)
    float*    accF;                 // EPI 1: += roww[row] * (c+bias)
    const float* roww; int rstride; // EPI 1
    const float* op;  const float* om;   // EPI 3/4: out_plus / out_minus
    float* mrF; _Float16* mrH;           // EPI 3: modified repulsion out
    const float* pt; const float* tscale; float* outF; // EPI 4: final output
};

template <int EPI>
__device__ __forceinline__ void gemm_epi(const GemmP& p, f8 acc, int row0, int col) {
    float bv = p.bias[col];
#pragma unroll
    for (int r = 0; r < 8; ++r) {
        int row = row0 + r;
        float c = acc[r] + bv;
        size_t idx = (size_t)row * p.N + col;
        if (EPI == 0) {                       // ReLU -> f16
            c = fmaxf(c, 0.0f);
            p.outH[idx] = (_Float16)c;
        } else if (EPI == 1) {                // MoE gate-weighted accumulate
            float w = p.roww[(size_t)row * p.rstride];
            p.accF[idx] += w * c;
        } else if (EPI == 2) {                // tanh -> f16
            p.outH[idx] = (_Float16)tanhf(c);
        } else if (EPI == 3) {                // modrep = (op-om) + influence
            float v = (p.op[idx] - p.om[idx]) + c;
            p.mrF[idx] = v;
            p.mrH[idx] = (_Float16)v;
        } else {                              // final output
            float fd = tanhf(c);
            float eq = 0.5f * (p.op[idx] + p.om[idx]);
            float fs = (*p.tscale) * sqrtf(p.pt[row] + 1e-8f);
            p.outF[idx] = eq + fs * fd;
        }
    }
}

template <int EPI>
__global__ __launch_bounds__(256) void gemm_k(GemmP p) {
    __shared__ _Float16 As[128 * 40];   // [m][k], row stride 40 halves (padded)
    __shared__ _Float16 Bs[128 * 40];   // [n][k], row stride 40 halves (padded)

    const int tid  = threadIdx.x;
    const int lane = tid & 31;
    const int wid  = tid >> 5;
    const int wm   = wid >> 1;          // 0..3  (M band of 32)
    const int wn   = wid & 1;           // 0..1  (N band of 64)
    const int half = lane >> 4;         // 0/1
    const int nl   = lane & 15;

    const int bn = blockIdx.x * 128;
    const int bm = blockIdx.y * 128;

    // A staging (async direct to LDS): 2x b128 per thread
    const int am  = tid >> 1;           // 0..127 (A row)
    const int akq = (tid & 1) << 4;     // 0 / 16 (A k-half)
    // B staging: row kb, two 8-column chunks at n8 and n8+64
    const int bk  = tid >> 3;           // 0..31  (B k-row)
    const int bn8 = (tid & 7) << 3;     // 0..56

    const _Float16* gA = p.A + (size_t)(bm + am) * p.lda + akq;
    const float*    gB = p.B + (size_t)bk * p.ldb + bn + bn8;
    const unsigned  aLds = (unsigned)(unsigned long long)&As[am * 40 + akq];

    f8 acc[2][4] = {};

    for (int k0 = 0; k0 < p.K; k0 += 32) {
        // ---- stage A tile (f16, 128x32) via async copy to LDS ----
        {
            unsigned long long g = (unsigned long long)(gA + k0);
            asm volatile("global_load_async_to_lds_b128 %0, %1, off\n\t"
                         "global_load_async_to_lds_b128 %0, %1, off offset:16"
                         :: "v"(aLds), "v"(g) : "memory");
        }
        // ---- stage B tile (f32 -> f16, transposed to [n][k]) ----
#pragma unroll
        for (int c = 0; c < 2; ++c) {
            const float* s = gB + (size_t)k0 * p.ldb + c * 64;
            float4 f0 = *(const float4*)s;
            float4 f1 = *(const float4*)(s + 4);
            _Float16 hv[8] = {(_Float16)f0.x, (_Float16)f0.y, (_Float16)f0.z, (_Float16)f0.w,
                              (_Float16)f1.x, (_Float16)f1.y, (_Float16)f1.z, (_Float16)f1.w};
#pragma unroll
            for (int i = 0; i < 8; ++i) Bs[(bn8 + c * 64 + i) * 40 + bk] = hv[i];
        }
        if (k0 + 32 < p.K) {
            __builtin_prefetch(gB + (size_t)(k0 + 32) * p.ldb, 0, 1);
            __builtin_prefetch(gB + (size_t)(k0 + 32) * p.ldb + 64, 0, 1);
        }
        asm volatile("s_wait_asynccnt 0x0" ::: "memory");
        __syncthreads();

        // ---- fragments per ISA f16 VGPR layouts ----
        const int ar = (wm << 5) + nl;
        const h8* qa0 = (const h8*)&As[ar * 40        + (half << 3)];
        const h8* qa1 = (const h8*)&As[(ar + 16) * 40 + (half << 3)];
        h16 af0 = cat16(qa0[0], qa0[2]);   // K = h*8..h*8+7, 16+h*8..16+h*8+7
        h16 af1 = cat16(qa1[0], qa1[2]);

        h16 bf[4];
#pragma unroll
        for (int t = 0; t < 4; ++t) {
            const int bc = (wn << 6) + (t << 4) + nl;
            const h8* qb = (const h8*)&Bs[bc * 40 + (half << 3)];
            bf[t] = cat16(qb[0], qb[2]);   // K = h*8..h*8+7, 16+h*8..16+h*8+7
        }

#pragma unroll
        for (int t = 0; t < 4; ++t) {
            acc[0][t] = wmma16(af0, bf[t], acc[0][t]);
            acc[1][t] = wmma16(af1, bf[t], acc[1][t]);
        }
        __syncthreads();
    }

    const int row0 = bm + (wm << 5) + (half << 3);
    const int col0 = bn + (wn << 6) + nl;
#pragma unroll
    for (int t = 0; t < 4; ++t) {
        gemm_epi<EPI>(p, acc[0][t], row0,      col0 + (t << 4));
        gemm_epi<EPI>(p, acc[1][t], row0 + 16, col0 + (t << 4));
    }
}

// ---------------------------------------------------------------------------
// Small helper kernels
// ---------------------------------------------------------------------------
__global__ void k_zero(float* a, size_t n) {
    size_t i = (size_t)blockIdx.x * blockDim.x + threadIdx.x;
    if (i < n) a[i] = 0.0f;
}

__global__ void k_cvt_f16(const float* in, _Float16* out, size_t n) {
    size_t i = (size_t)blockIdx.x * blockDim.x + threadIdx.x;
    if (i < n) out[i] = (_Float16)in[i];
}

// top-4-of-12 gating (one block per batch row)
__global__ __launch_bounds__(128) void k_gate_a(const float* x, const float* gw,
                                                const float* gb, float* wa) {
    __shared__ float red[128 * N_EXP_A];
    __shared__ float logit[N_EXP_A];
    const int row = blockIdx.x, tid = threadIdx.x;
    float p[N_EXP_A];
#pragma unroll
    for (int e = 0; e < N_EXP_A; ++e) p[e] = 0.0f;
    for (int i = tid; i < IN_DIM; i += 128) {
        float xv = x[(size_t)row * IN_DIM + i];
#pragma unroll
        for (int e = 0; e < N_EXP_A; ++e) p[e] += xv * gw[i * N_EXP_A + e];
    }
#pragma unroll
    for (int e = 0; e < N_EXP_A; ++e) red[tid * N_EXP_A + e] = p[e];
    __syncthreads();
    if (tid < N_EXP_A) {
        float s = 0.0f;
        for (int t = 0; t < 128; ++t) s += red[t * N_EXP_A + tid];
        logit[tid] = s + gb[tid];
    }
    __syncthreads();
    if (tid == 0) {
        float lv[N_EXP_A]; bool used[N_EXP_A];
#pragma unroll
        for (int e = 0; e < N_EXP_A; ++e) { lv[e] = logit[e]; used[e] = false; }
        float tv[TOPK_A]; int ti[TOPK_A];
        for (int j = 0; j < TOPK_A; ++j) {
            float best = -3.4e38f; int bi = 0;
            for (int e = 0; e < N_EXP_A; ++e)
                if (!used[e] && lv[e] > best) { best = lv[e]; bi = e; }
            used[bi] = true; tv[j] = best; ti[j] = bi;
        }
        float m = tv[0], sum = 0.0f, ev[TOPK_A];
        for (int j = 0; j < TOPK_A; ++j) { ev[j] = expf(tv[j] - m); sum += ev[j]; }
        for (int e = 0; e < N_EXP_A; ++e) wa[(size_t)row * N_EXP_A + e] = 0.0f;
        for (int j = 0; j < TOPK_A; ++j) wa[(size_t)row * N_EXP_A + ti[j]] = ev[j] / sum;
    }
}

// dense softmax-of-6 gating + per-row entropy
__global__ __launch_bounds__(128) void k_gate_g(const float* x, const float* gw,
                                                const float* gb, float* wg, float* hrow) {
    __shared__ float red[128 * N_EXP_G];
    __shared__ float logit[N_EXP_G];
    const int row = blockIdx.x, tid = threadIdx.x;
    float p[N_EXP_G];
#pragma unroll
    for (int e = 0; e < N_EXP_G; ++e) p[e] = 0.0f;
    for (int i = tid; i < IN_DIM; i += 128) {
        float xv = x[(size_t)row * IN_DIM + i];
#pragma unroll
        for (int e = 0; e < N_EXP_G; ++e) p[e] += xv * gw[i * N_EXP_G + e];
    }
#pragma unroll
    for (int e = 0; e < N_EXP_G; ++e) red[tid * N_EXP_G + e] = p[e];
    __syncthreads();
    if (tid < N_EXP_G) {
        float s = 0.0f;
        for (int t = 0; t < 128; ++t) s += red[t * N_EXP_G + tid];
        logit[tid] = s + gb[tid];
    }
    __syncthreads();
    if (tid == 0) {
        float m = -3.4e38f;
        for (int e = 0; e < N_EXP_G; ++e) m = fmaxf(m, logit[e]);
        float sum = 0.0f, ev[N_EXP_G];
        for (int e = 0; e < N_EXP_G; ++e) { ev[e] = expf(logit[e] - m); sum += ev[e]; }
        float h = 0.0f;
        for (int e = 0; e < N_EXP_G; ++e) {
            float w = ev[e] / sum;
            wg[(size_t)row * N_EXP_G + e] = w;
            h -= w * logf(w + 1e-8f);
        }
        hrow[row] = h;
    }
}

// pt[row] = sum_col (a-b)^2  (b may be null -> a^2)
__global__ __launch_bounds__(256) void k_rowsq(const float* a, const float* b, float* pt) {
    __shared__ float s[256];
    const int row = blockIdx.x, tid = threadIdx.x;
    float acc = 0.0f;
    for (int c = tid; c < OUT_DIM; c += 256) {
        size_t idx = (size_t)row * OUT_DIM + c;
        float v = b ? (a[idx] - b[idx]) : a[idx];
        acc += v * v;
    }
    s[tid] = acc;
    __syncthreads();
    for (int off = 128; off > 0; off >>= 1) {
        if (tid < off) s[tid] += s[tid + off];
        __syncthreads();
    }
    if (tid == 0) pt[row] = s[0];
}

// deterministic mean of n values -> *out
__global__ __launch_bounds__(256) void k_mean(const float* v, int n, float* out) {
    __shared__ float s[256];
    const int tid = threadIdx.x;
    float acc = 0.0f;
    for (int i = tid; i < n; i += 256) acc += v[i];
    s[tid] = acc;
    __syncthreads();
    for (int off = 128; off > 0; off >>= 1) {
        if (tid < off) s[tid] += s[tid + off];
        __syncthreads();
    }
    if (tid == 0) *out = s[0] / (float)n;
}

// hidden = tanh(self_input @ sm_w1 + sm_b1), self_input = [pt, delta, step/3]
__global__ __launch_bounds__(256) void k_selfhidden(const float* pt, const float* tprev,
                                                    int step, const float* w1,
                                                    const float* b1, _Float16* outH) {
    const int row = blockIdx.x, tid = threadIdx.x;
    float ts = pt[row];
    float td = (step == 0) ? 0.0f : (ts - *tprev);
    float st = (float)step / 3.0f;
    for (int h = tid; h < HID_DIM; h += 256) {
        float v = ts * w1[h] + td * w1[HID_DIM + h] + st * w1[2 * HID_DIM + h] + b1[h];
        outH[(size_t)row * HID_DIM + h] = (_Float16)tanhf(v);
    }
}

// aux_loss scalar:  (h_mean - H)^2 + 0.001 * mean|t[i+1]-t[i]|
__global__ void k_aux(const float* scal, float* out) {
    float ent = scal[0] - H_TARGET;
    ent = ent * ent;
    float conv = (fabsf(scal[2] - scal[1]) + fabsf(scal[3] - scal[2]) +
                  fabsf(scal[4] - scal[3])) / 3.0f;
    *out = ent + 0.001f * conv;
}

// ---------------------------------------------------------------------------
// Host side
// ---------------------------------------------------------------------------
static void gemm_launch(int epi, const GemmP& p, hipStream_t st) {
    dim3 g(p.N / 128, p.M / 128), b(256);
    switch (epi) {
        case 0: gemm_k<0><<<g, b, 0, st>>>(p); break;
        case 1: gemm_k<1><<<g, b, 0, st>>>(p); break;
        case 2: gemm_k<2><<<g, b, 0, st>>>(p); break;
        case 3: gemm_k<3><<<g, b, 0, st>>>(p); break;
        default: gemm_k<4><<<g, b, 0, st>>>(p); break;
    }
}

extern "C" void kernel_launch(void* const* d_in, const int* in_sizes, int n_in,
                              void* d_out, int out_size, void* d_ws, size_t ws_size,
                              hipStream_t stream) {
    (void)in_sizes; (void)n_in; (void)out_size; (void)ws_size;
    const float* x     = (const float*)d_in[0];
    const float* ga_w  = (const float*)d_in[1];
    const float* ga_b  = (const float*)d_in[2];
    const float* wa1   = (const float*)d_in[3];
    const float* ba1   = (const float*)d_in[4];
    const float* wa2   = (const float*)d_in[5];
    const float* ba2   = (const float*)d_in[6];
    const float* gg_w  = (const float*)d_in[7];
    const float* gg_b  = (const float*)d_in[8];
    const float* wg1   = (const float*)d_in[9];
    const float* bg1   = (const float*)d_in[10];
    const float* wg2   = (const float*)d_in[11];
    const float* bg2   = (const float*)d_in[12];
    const float* sm_w1 = (const float*)d_in[13];
    const float* sm_b1 = (const float*)d_in[14];
    const float* sm_w2 = (const float*)d_in[15];
    const float* sm_b2 = (const float*)d_in[16];
    const float* si_w  = (const float*)d_in[17];
    const float* si_b  = (const float*)d_in[18];
    const float* ft_w  = (const float*)d_in[19];
    const float* ft_b  = (const float*)d_in[20];
    const float* tsc   = (const float*)d_in[21];
    float* out = (float*)d_out;

    // ---- workspace layout ----
    char* ws = (char*)d_ws;
    const size_t SZ_BO = (size_t)BATCH * OUT_DIM * 4;      // 8 MB
    float*    out_plus  = (float*)(ws);                    // [B,O]
    float*    out_minus = (float*)(ws + SZ_BO);            // [B,O] (contiguous after)
    float*    modrepF   = (float*)(ws + 2 * SZ_BO);        // [B,O]
    _Float16* Xh        = (_Float16*)(ws + 3 * SZ_BO);     // [B,I] f16 (reused as self_state)
    _Float16* Hh        = (_Float16*)(ws + 3 * SZ_BO + (size_t)BATCH * IN_DIM * 2); // [B,H] f16
    char* q = ws + 3 * SZ_BO + (size_t)BATCH * IN_DIM * 2 + (size_t)BATCH * HID_DIM * 2;
    _Float16* mrH = (_Float16*)q;            q += (size_t)BATCH * OUT_DIM * 2;  // [B,O] f16
    float* wA   = (float*)q;                 q += (size_t)BATCH * N_EXP_A * 4;
    float* wG   = (float*)q;                 q += (size_t)BATCH * N_EXP_G * 4;
    float* pt   = (float*)q;                 q += (size_t)BATCH * 4;
    float* hrow = (float*)q;                 q += (size_t)BATCH * 4;
    float* scal = (float*)q;  // [0]=h_mean, [1..4]=t0..t3

    // 1) zero accumulators (out_plus & out_minus contiguous)
    {
        size_t n = 2 * (size_t)BATCH * OUT_DIM;
        k_zero<<<dim3((unsigned)((n + 255) / 256)), 256, 0, stream>>>(out_plus, n);
    }
    // 2) x -> f16
    {
        size_t n = (size_t)BATCH * IN_DIM;
        k_cvt_f16<<<dim3((unsigned)((n + 255) / 256)), 256, 0, stream>>>(x, Xh, n);
    }
    // 3) gates
    k_gate_a<<<BATCH, 128, 0, stream>>>(x, ga_w, ga_b, wA);
    k_gate_g<<<BATCH, 128, 0, stream>>>(x, gg_w, gg_b, wG, hrow);
    k_mean<<<1, 256, 0, stream>>>(hrow, BATCH, &scal[0]);

    // 4) EngineA experts (serialized -> plain accumulate is race-free)
    for (int e = 0; e < N_EXP_A; ++e) {
        GemmP p1 = {};
        p1.A = Xh; p1.B = wa1 + (size_t)e * IN_DIM * HID_DIM; p1.bias = ba1 + (size_t)e * HID_DIM;
        p1.M = BATCH; p1.N = HID_DIM; p1.K = IN_DIM; p1.lda = IN_DIM; p1.ldb = HID_DIM;
        p1.outH = Hh;
        gemm_launch(0, p1, stream);
        GemmP p2 = {};
        p2.A = Hh; p2.B = wa2 + (size_t)e * HID_DIM * OUT_DIM; p2.bias = ba2 + (size_t)e * OUT_DIM;
        p2.M = BATCH; p2.N = OUT_DIM; p2.K = HID_DIM; p2.lda = HID_DIM; p2.ldb = OUT_DIM;
        p2.accF = out_plus; p2.roww = wA + e; p2.rstride = N_EXP_A;
        gemm_launch(1, p2, stream);
    }
    // 5) EngineG experts
    for (int e = 0; e < N_EXP_G; ++e) {
        GemmP p1 = {};
        p1.A = Xh; p1.B = wg1 + (size_t)e * IN_DIM * HID_DIM; p1.bias = bg1 + (size_t)e * HID_DIM;
        p1.M = BATCH; p1.N = HID_DIM; p1.K = IN_DIM; p1.lda = IN_DIM; p1.ldb = HID_DIM;
        p1.outH = Hh;
        gemm_launch(0, p1, stream);
        GemmP p2 = {};
        p2.A = Hh; p2.B = wg2 + (size_t)e * HID_DIM * OUT_DIM; p2.bias = bg2 + (size_t)e * OUT_DIM;
        p2.M = BATCH; p2.N = OUT_DIM; p2.K = HID_DIM; p2.lda = HID_DIM; p2.ldb = OUT_DIM;
        p2.accF = out_minus; p2.roww = wG + e; p2.rstride = N_EXP_G;
        gemm_launch(1, p2, stream);
    }

    // 6) initial tension from repulsion = out_plus - out_minus
    k_rowsq<<<BATCH, 256, 0, stream>>>(out_plus, out_minus, pt);
    k_mean<<<1, 256, 0, stream>>>(pt, BATCH, &scal[1]);

    // 7) self-referential loop (n_steps = 3)
    for (int s = 0; s < 3; ++s) {
        k_selfhidden<<<BATCH, 256, 0, stream>>>(pt, &scal[1 + s], s, sm_w1, sm_b1, Hh);
        GemmP p3 = {};   // self_state = tanh(hidden @ sm_w2 + sm_b2)   (f16 into Xh)
        p3.A = Hh; p3.B = sm_w2; p3.bias = sm_b2;
        p3.M = BATCH; p3.N = OUT_DIM; p3.K = HID_DIM; p3.lda = HID_DIM; p3.ldb = OUT_DIM;
        p3.outH = Xh;
        gemm_launch(2, p3, stream);
        GemmP p4 = {};   // modrep = (out_plus - out_minus) + self_state @ si_w + si_b
        p4.A = Xh; p4.B = si_w; p4.bias = si_b;
        p4.M = BATCH; p4.N = OUT_DIM; p4.K = OUT_DIM; p4.lda = OUT_DIM; p4.ldb = OUT_DIM;
        p4.op = out_plus; p4.om = out_minus; p4.mrF = modrepF; p4.mrH = mrH;
        gemm_launch(3, p4, stream);
        k_rowsq<<<BATCH, 256, 0, stream>>>(modrepF, nullptr, pt);
        k_mean<<<1, 256, 0, stream>>>(pt, BATCH, &scal[2 + s]);
    }

    // 8) output = 0.5*(op+om) + tscale*sqrt(pt+1e-8)*tanh(modrep @ ft_w + ft_b)
    {
        GemmP p5 = {};
        p5.A = mrH; p5.B = ft_w; p5.bias = ft_b;
        p5.M = BATCH; p5.N = OUT_DIM; p5.K = OUT_DIM; p5.lda = OUT_DIM; p5.ldb = OUT_DIM;
        p5.op = out_plus; p5.om = out_minus; p5.pt = pt; p5.tscale = tsc; p5.outF = out;
        gemm_launch(4, p5, stream);
    }
    // 9) aux loss scalar
    k_aux<<<1, 1, 0, stream>>>(scal, out + (size_t)BATCH * OUT_DIM);
}
